// SelfAttention_7438883356871
// MI455X (gfx1250) — matrix operand order repaired and checked by reading, not compiled, not run on hardware
//
#include <hip/hip_runtime.h>

typedef __attribute__((ext_vector_type(16))) _Float16 v16h;
typedef __attribute__((ext_vector_type(8)))  float    v8f;

#define B_      16
#define N_      2048
#define D_      64
#define QT      16            // query rows per workgroup
#define KT      16            // key cols per score tile
#define WAVES   4
#define THREADS (WAVES * 32)
#define NKT     (N_ / KT)     // 128 key tiles

__global__ __launch_bounds__(THREADS)
void attn_wmma_kernel(const float* __restrict__ q,
                      const float* __restrict__ k,
                      const float* __restrict__ v,
                      const float* __restrict__ scale_p,
                      float* __restrict__ ctx,
                      float* __restrict__ attn)
{
    extern __shared__ float P[];            // [QT][N_] scores -> probabilities (128 KB)

    const int tid  = threadIdx.x;
    const int wave = tid >> 5;
    const int lane = tid & 31;
    const int half = lane >> 4;             // 0: lanes 0-15, 1: lanes 16-31
    const int l16  = lane & 15;

    const int b     = blockIdx.x / (N_ / QT);
    const int qt    = blockIdx.x % (N_ / QT);
    const int qrow0 = qt * QT;

    const float scale = scale_p[0];

    // ---- Q tile as two 16x32 f16 A-fragments (D=64 -> 2 chunks of K=32).
    // A layout: lane holds row m=l16; VGPR vi holds K = c*32 + 16*half + 2*vi.
    v16h aq[2];
    {
        const float* qrow = q + ((size_t)b * N_ + qrow0 + l16) * D_;
#pragma unroll
        for (int c = 0; c < 2; ++c)
#pragma unroll
            for (int vi = 0; vi < 8; ++vi) {
                int d = c * 32 + half * 16 + 2 * vi;
                aq[c][2 * vi]     = (_Float16)qrow[d];
                aq[c][2 * vi + 1] = (_Float16)qrow[d + 1];
            }
    }

    // ---- Phase 1: scores = scale * (Q K^T) into LDS; each wave does 32 key tiles.
    for (int kt = wave * (NKT / WAVES); kt < (wave + 1) * (NKT / WAVES); ++kt) {
        // B layout (32x16): lane holds col n=l16; VGPR vi holds K = c*32 + 16*half + 2*vi.
        v16h bk[2];
        const float* krow = k + ((size_t)b * N_ + kt * KT + l16) * D_;
#pragma unroll
        for (int c = 0; c < 2; ++c)
#pragma unroll
            for (int vi = 0; vi < 8; ++vi) {
                int d = c * 32 + half * 16 + 2 * vi;
                bk[c][2 * vi]     = (_Float16)krow[d];
                bk[c][2 * vi + 1] = (_Float16)krow[d + 1];
            }
        v8f acc = {};
        acc = __builtin_amdgcn_wmma_f32_16x16x32_f16(false, aq[0], false, bk[0],
                                                     (short)0, acc, false, false);
        acc = __builtin_amdgcn_wmma_f32_16x16x32_f16(false, aq[1], false, bk[1],
                                                     (short)0, acc, false, false);
        // C/D layout: col n=l16; VGPR r is row m = r + 8*half.
#pragma unroll
        for (int r = 0; r < 8; ++r) {
            int m = r + 8 * half;
            P[m * N_ + kt * KT + l16] = acc[r] * scale;
        }
    }
    __syncthreads();

    // ---- Phase 2: row softmax over N=2048; single exp per element.
    for (int m = wave * 4; m < wave * 4 + 4; ++m) {
        float mx = -3.0e38f;
        for (int i = lane; i < N_; i += 32) mx = fmaxf(mx, P[m * N_ + i]);
#pragma unroll
        for (int off = 16; off > 0; off >>= 1) mx = fmaxf(mx, __shfl_xor(mx, off, 32));

        // exp in place + partial sums (each lane rereads its own writes; same-wave
        // DS ops are in-order, so no extra sync needed between the two sweeps)
        float sum = 0.0f;
        for (int i = lane; i < N_; i += 32) {
            float e = __expf(P[m * N_ + i] - mx);
            P[m * N_ + i] = e;
            sum += e;
        }
#pragma unroll
        for (int off = 16; off > 0; off >>= 1) sum += __shfl_xor(sum, off, 32);
        const float inv = 1.0f / sum;

        float* arow = attn + ((size_t)b * N_ + qrow0 + m) * N_;
        for (int i = lane; i < N_; i += 32) {
            float p = P[m * N_ + i] * inv;
            P[m * N_ + i] = p;        // keep normalized probs in LDS for PV
            arow[i] = p;              // coalesced 128B stores
        }
    }
    __syncthreads();

    // ---- Phase 3: context = P @ V; wave owns output columns [wave*16, wave*16+16).
    {
        const int n0 = wave * 16;
        v8f acc = {};
        for (int kk = 0; kk < N_ / 32; ++kk) {
            v16h ap, bv;
#pragma unroll
            for (int vi = 0; vi < 8; ++vi) {
                int kA = kk * 32 + half * 16 + 2 * vi;
                ap[2 * vi]     = (_Float16)P[l16 * N_ + kA];
                ap[2 * vi + 1] = (_Float16)P[l16 * N_ + kA + 1];
                int kB = kk * 32 + half * 16 + 2 * vi;
                const float* vr = v + ((size_t)b * N_ + kB) * D_ + n0 + l16;
                bv[2 * vi]     = (_Float16)vr[0];
                bv[2 * vi + 1] = (_Float16)vr[D_];
            }
            acc = __builtin_amdgcn_wmma_f32_16x16x32_f16(false, ap, false, bv,
                                                         (short)0, acc, false, false);
        }
#pragma unroll
        for (int r = 0; r < 8; ++r) {
            int m = r + 8 * half;
            ctx[((size_t)b * N_ + qrow0 + m) * D_ + n0 + l16] = acc[r];
        }
    }
}

extern "C" void kernel_launch(void* const* d_in, const int* in_sizes, int n_in,
                              void* d_out, int out_size, void* d_ws, size_t ws_size,
                              hipStream_t stream) {
    const float* q = (const float*)d_in[0];
    const float* k = (const float*)d_in[1];
    const float* v = (const float*)d_in[2];
    const float* s = (const float*)d_in[3];

    float* ctx  = (float*)d_out;                              // [B,N,D] first
    float* attn = (float*)d_out + (size_t)B_ * N_ * D_;       // [B,N,N] second

    const size_t smem = (size_t)QT * N_ * sizeof(float);      // 128 KB (<= 320 KB/WGP)
    hipFuncSetAttribute((const void*)attn_wmma_kernel,
                        hipFuncAttributeMaxDynamicSharedMemorySize, (int)smem);

    dim3 grid(B_ * (N_ / QT));    // 2048 workgroups
    dim3 block(THREADS);          // 4 waves (wave32)
    attn_wmma_kernel<<<grid, block, smem, stream>>>(q, k, v, s, ctx, attn);
}
